// MultiWayContextualAttention_79688823210166
// MI455X (gfx1250) — matrix operand order repaired and checked by reading, not compiled, not run on hardware
//
#include <hip/hip_runtime.h>
#include <hip/hip_bf16.h>

// Problem constants (match reference)
#define BB     8
#define NSEQ   1024
#define DMODEL 768
#define HH     8
#define EE     6
#define DHD    96
#define DFF    3072
#define NTOK   (BB * NSEQ)   // 8192 tokens

typedef unsigned short u16;
typedef unsigned int   u32;
typedef unsigned long long u64;
typedef __attribute__((ext_vector_type(16))) __bf16 v16bf;
typedef __attribute__((ext_vector_type(8)))  float  v8f;
typedef __attribute__((ext_vector_type(4)))  unsigned int v4u;
typedef __attribute__((ext_vector_type(8)))  int v8i;
typedef __attribute__((ext_vector_type(4)))  int v4i;

#if defined(__has_builtin)
#if __has_builtin(__builtin_amdgcn_tensor_load_to_lds)
#define USE_TDM 1
#endif
#endif
#ifndef USE_TDM
#define USE_TDM 0
#endif

// ---------------- bf16 helpers ----------------
__device__ __forceinline__ u16 f2bf(float f) {
  u32 u = __builtin_bit_cast(u32, f);
  u32 r = u + 0x7FFFu + ((u >> 16) & 1u);   // round-to-nearest-even
  return (u16)(r >> 16);
}

union FragU {
  v16bf v;
  u16   u[16];
};

// A fragment (16x32 bf16, M x K): lane<16 -> M=lane, K = k0+[0..7], k0+[16..23]
//                                 lane>=16 -> M=lane-16, K = k0+[8..15], k0+[24..31]
__device__ __forceinline__ v16bf load_frag_a_bf16(const u16* base, int ld) {
  const int lane = threadIdx.x & 31;
  const int half = lane >> 4;
  const u16* row = base + (size_t)(lane & 15) * ld + half * 8;
  FragU f;
  *(uint4*)(&f.u[0]) = *(const uint4*)(row);
  *(uint4*)(&f.u[8]) = *(const uint4*)(row + 16);
  return f.v;
}

// B fragment (32x16 bf16, K x N), source stored N-major (row = N, K contiguous):
// lane<16 -> N=lane, K=k0+[0..7], k0+[16..23]; lane>=16 -> N=lane-16, K=k0+[8..15], k0+[24..31]
__device__ __forceinline__ v16bf load_frag_b_bf16(const u16* base, int ld) {
  const int lane = threadIdx.x & 31;
  const int half = lane >> 4;
  const u16* row = base + (size_t)(lane & 15) * ld + half * 8;
  FragU f;
  *(uint4*)(&f.u[0]) = *(const uint4*)(row);
  *(uint4*)(&f.u[8]) = *(const uint4*)(row + 16);
  return f.v;
}

// A fragment from fp32 source (convert on load) — used for softmaxed attention probs
__device__ __forceinline__ v16bf load_frag_a_f32(const float* base, int ld) {
  const int lane = threadIdx.x & 31;
  const int half = lane >> 4;
  const float* row = base + (size_t)(lane & 15) * ld + half * 8;
  float4 x0 = *(const float4*)(row);
  float4 x1 = *(const float4*)(row + 4);
  float4 x2 = *(const float4*)(row + 16);
  float4 x3 = *(const float4*)(row + 20);
  FragU f;
  f.u[0]  = f2bf(x0.x); f.u[1]  = f2bf(x0.y); f.u[2]  = f2bf(x0.z); f.u[3]  = f2bf(x0.w);
  f.u[4]  = f2bf(x1.x); f.u[5]  = f2bf(x1.y); f.u[6]  = f2bf(x1.z); f.u[7]  = f2bf(x1.w);
  f.u[8]  = f2bf(x2.x); f.u[9]  = f2bf(x2.y); f.u[10] = f2bf(x2.z); f.u[11] = f2bf(x2.w);
  f.u[12] = f2bf(x3.x); f.u[13] = f2bf(x3.y); f.u[14] = f2bf(x3.z); f.u[15] = f2bf(x3.w);
  return f.v;
}

__device__ __forceinline__ v8f wmma_bf16(v16bf a, v16bf b, v8f c) {
  // 8 args: (neg_a, A, neg_b, B, c_mod, C, reuse_a, reuse_b)
  return __builtin_amdgcn_wmma_f32_16x16x32_bf16(false, a, false, b, (short)0, c,
                                                 false, false);
}

__device__ __forceinline__ void async_copy_b128(u32 lds_addr, const u16* gsrc) {
  // GLOBAL_LOAD_ASYNC_TO_LDS_B128: VDST = LDS byte address VGPR, VADDR = 64-bit global addr
  asm volatile("global_load_async_to_lds_b128 %0, %1, off"
               :: "v"(lds_addr), "v"((u64)(uintptr_t)gsrc) : "memory");
}

__device__ __forceinline__ void wait_async(int n) {
  if (n == 0)      asm volatile("s_wait_asynccnt 0x0" ::: "memory");
  else if (n == 2) asm volatile("s_wait_asynccnt 0x2" ::: "memory");
  else             asm volatile("s_wait_asynccnt 0x4" ::: "memory");
}

// ===================== LDS-staged GEMM: C = A(MxK) * Bt(N-major,NxK) + bias (+res) =====================
// block = 256 threads = 8 waves. Block tile 128(M) x 128(N), k-step 32.
// A tile staged by all waves via ASYNC global->LDS copies; B tile staged by the
// Tensor Data Mover (2D D#, TENSORcnt) when available, else async copies.
// Each wave computes one 16-row strip against eight 16-col B tiles (8 accumulators).
template <bool OUT_BF16, bool RELU, bool MASKED>
__global__ void gemm_kernel(const u16* __restrict__ A, int lda,
                            const u16* __restrict__ Bt, int ldb,
                            const float* __restrict__ bias,
                            const float* __restrict__ residual, int ldr,
                            void* __restrict__ Cout, int ldc, int K,
                            const int* __restrict__ wsel, int expert) {
  __shared__ __align__(16) u16 smA[2][128 * 32];   // 2 x 8 KB
  __shared__ __align__(16) u16 smB[2][128 * 32];   // 2 x 8 KB
  const int tid  = threadIdx.x;
  const int wave = tid >> 5;
  const int lane = tid & 31;
  const int m0 = blockIdx.y * 128;
  const int n0 = blockIdx.x * 128;
  const int Ntot = (int)(gridDim.x << 7);

  auto stage = [&](int buf, int k0) {
    // A: 128 rows x 64B = 512 x 16B chunks, 2 per thread (ASYNCcnt path)
#pragma unroll
    for (int i = 0; i < 2; ++i) {
      const int c = tid + i * 256;
      const int row = c >> 2, col = (c & 3) * 8;
      async_copy_b128((u32)(uintptr_t)&smA[buf][row * 32 + col],
                      A + (size_t)(m0 + row) * lda + k0 + col);
    }
#if USE_TDM
    // B: one TDM 2D tile descriptor, issued by wave 0 only (TENSORcnt path).
    // D# group0: count=1 | lds_addr | global_addr(57b) | type=2
    // D# group1: data_size=2B, tensor_dim0=K, tensor_dim1=Ntot,
    //            tile_dim0=32, tile_dim1=128, tensor_dim0_stride=ldb
    if (wave == 0) {
      const u64 ga = (u64)(uintptr_t)(Bt + (size_t)n0 * ldb + k0);
      const u32 la = (u32)(uintptr_t)&smB[buf][0];
      v4u g0 = { 1u, la, (u32)ga,
                 (u32)((ga >> 32) & 0x1FFFFFFu) | (2u << 30) };
      v8i g1 = { (int)(1u << 16),
                 (int)(((u32)K & 0xFFFFu) << 16),
                 (int)(((u32)K >> 16) | (((u32)Ntot & 0xFFFFu) << 16)),
                 (int)(((u32)Ntot >> 16) | (32u << 16)),
                 128,
                 (int)(u32)ldb,
                 0, 0 };
      v4i gz4 = { 0, 0, 0, 0 };
      v8i gz8 = { 0, 0, 0, 0, 0, 0, 0, 0 };
      __builtin_amdgcn_tensor_load_to_lds(g0, g1, gz4, gz4, gz8, 0);
    }
#else
    // B fallback: async copies, 512 chunks, 2 per thread
#pragma unroll
    for (int i = 0; i < 2; ++i) {
      const int c = tid + i * 256;
      const int row = c >> 2, col = (c & 3) * 8;
      async_copy_b128((u32)(uintptr_t)&smB[buf][row * 32 + col],
                      Bt + (size_t)(n0 + row) * ldb + k0 + col);
    }
#endif
  };

  auto frag_a = [&](int buf) -> v16bf {
    const int half = lane >> 4;
    const u16* row = &smA[buf][(wave * 16 + (lane & 15)) * 32 + half * 8];
    FragU f;
    *(uint4*)(&f.u[0]) = *(const uint4*)(row);
    *(uint4*)(&f.u[8]) = *(const uint4*)(row + 16);
    return f.v;
  };
  auto frag_b = [&](int buf, int j) -> v16bf {
    const int half = lane >> 4;
    const u16* row = &smB[buf][(j * 16 + (lane & 15)) * 32 + half * 8];
    FragU f;
    *(uint4*)(&f.u[0]) = *(const uint4*)(row);
    *(uint4*)(&f.u[8]) = *(const uint4*)(row + 16);
    return f.v;
  };

  v8f acc[8] = {{}, {}, {}, {}, {}, {}, {}, {}};
  const int KT = K >> 5;
  stage(0, 0);
  for (int kt = 0; kt < KT; ++kt) {
    const int cur = kt & 1;
    if (kt + 1 < KT) {
      stage(cur ^ 1, (kt + 1) << 5);
#if USE_TDM
      wait_async(2);                               // A: 2 in flight for next stage
      if (wave == 0) __builtin_amdgcn_s_wait_tensorcnt(1);
#else
      wait_async(4);
#endif
    } else {
      wait_async(0);
#if USE_TDM
      if (wave == 0) __builtin_amdgcn_s_wait_tensorcnt(0);
#endif
    }
    __syncthreads();
    const v16bf a = frag_a(cur);
#pragma unroll
    for (int j = 0; j < 8; ++j)
      acc[j] = wmma_bf16(a, frag_b(cur, j), acc[j]);
    __syncthreads();   // all reads of buf `cur` done before it is restaged
  }

  const int nl = lane & 15;
  const int mb = (lane >> 4) * 8;
#pragma unroll
  for (int j = 0; j < 8; ++j) {
    const int gn = n0 + j * 16 + nl;
    const float bv = bias ? bias[gn] : 0.0f;
#pragma unroll
    for (int r = 0; r < 8; ++r) {
      const int gm = m0 + wave * 16 + mb + r;
      if (MASKED && wsel[gm] != expert) continue;
      float val = acc[j][r] + bv;
      if (residual) val += residual[(size_t)gm * ldr + gn];
      if (RELU) val = fmaxf(val, 0.0f);
      if (OUT_BF16) ((u16*)Cout)[(size_t)gm * ldc + gn] = f2bf(val);
      else          ((float*)Cout)[(size_t)gm * ldc + gn] = val;
    }
  }
}

// V projection with transposed (head-major) store: vT[(b*H+h)*DH + dh][token]
__global__ void gemm_vt_kernel(const u16* __restrict__ A, int lda,
                               const u16* __restrict__ Bt, int ldb,
                               const float* __restrict__ bias,
                               u16* __restrict__ vT, int K) {
  const int n0 = blockIdx.x * 16;
  const int m0 = (blockIdx.y * 8 + (threadIdx.x >> 5)) * 16;
  const u16* abase = A + (size_t)m0 * lda;
  const u16* bbase = Bt + (size_t)n0 * ldb;
  v8f c = {};
  for (int k0 = 0; k0 < K; k0 += 32) {
    __builtin_prefetch(abase + k0 + 128, 0, 0);
    v16bf af = load_frag_a_bf16(abase + k0, lda);
    v16bf bf = load_frag_b_bf16(bbase + k0, ldb);
    c = wmma_bf16(af, bf, c);
  }
  const int lane = threadIdx.x & 31;
  const int nl = lane & 15;
  const int mb = (lane >> 4) * 8;
  const int col = n0 + nl;                 // 0..D-1 ; tile never crosses a head (DH=96)
  const int h = col / DHD, dh = col % DHD;
  const float bv = bias[col];
#pragma unroll
  for (int r = 0; r < 8; ++r) {
    const int gm = m0 + mb + r;            // token id
    const int b = gm / NSEQ, t = gm % NSEQ;
    vT[((size_t)(b * HH + h) * DHD + dh) * NSEQ + t] = f2bf(c[r] + bv);
  }
}

// scores[b,h,q,k] = scale * Q . K  (K-dim = DH = 96)
__global__ void attn_scores_kernel(const u16* __restrict__ q,
                                   const u16* __restrict__ k,
                                   float* __restrict__ scores, float scale) {
  const int z = blockIdx.z, b = z / HH, h = z % HH;
  const int n0 = blockIdx.x * 16;
  const int m0 = (blockIdx.y * 8 + (threadIdx.x >> 5)) * 16;
  const u16* abase = q + (size_t)(b * NSEQ + m0) * DMODEL + h * DHD;
  const u16* bbase = k + (size_t)(b * NSEQ + n0) * DMODEL + h * DHD;
  v8f c = {};
#pragma unroll
  for (int k0 = 0; k0 < DHD; k0 += 32) {
    v16bf af = load_frag_a_bf16(abase + k0, DMODEL);
    v16bf bf = load_frag_b_bf16(bbase + k0, DMODEL);
    c = wmma_bf16(af, bf, c);
  }
  float* out = scores + (size_t)z * NSEQ * NSEQ;
  const int lane = threadIdx.x & 31;
  const int nl = lane & 15;
  const int mb = (lane >> 4) * 8;
#pragma unroll
  for (int r = 0; r < 8; ++r)
    out[(size_t)(m0 + mb + r) * NSEQ + n0 + nl] = c[r] * scale;
}

// out[b, q, h*DH+dh] = sum_k attn[b,h,q,k] * vT[b,h,dh,k]   (K-dim = NSEQ)
__global__ void attn_av_kernel(const float* __restrict__ attn,
                               const u16* __restrict__ vT,
                               u16* __restrict__ outb) {
  const int z = blockIdx.z, b = z / HH, h = z % HH;
  const int n0 = blockIdx.x * 16;    // dh tile
  const int m0 = (blockIdx.y * 8 + (threadIdx.x >> 5)) * 16;
  const float* abase = attn + (size_t)z * NSEQ * NSEQ + (size_t)m0 * NSEQ;
  const u16*   bbase = vT + ((size_t)z * DHD + n0) * NSEQ;
  v8f c = {};
  for (int k0 = 0; k0 < NSEQ; k0 += 32) {
    __builtin_prefetch(abase + k0 + 128, 0, 0);
    v16bf af = load_frag_a_f32(abase + k0, NSEQ);
    v16bf bf = load_frag_b_bf16(bbase + k0, NSEQ);
    c = wmma_bf16(af, bf, c);
  }
  const int lane = threadIdx.x & 31;
  const int nl = lane & 15;
  const int mb = (lane >> 4) * 8;
#pragma unroll
  for (int r = 0; r < 8; ++r)
    outb[(size_t)(b * NSEQ + m0 + mb + r) * DMODEL + h * DHD + n0 + nl] = f2bf(c[r]);
}

// ---------------- LayerNorm: fp32 in -> bf16 out ----------------
__global__ void ln_kernel(const float* __restrict__ x, const float* __restrict__ g,
                          const float* __restrict__ bta, u16* __restrict__ y, float eps) {
  const int row = blockIdx.x;
  const float* xr = x + (size_t)row * DMODEL;
  __shared__ float s1[256], s2[256];
  float a = 0.f, b2 = 0.f;
  for (int i = threadIdx.x; i < DMODEL; i += 256) {
    float v = xr[i];
    a += v; b2 += v * v;
  }
  s1[threadIdx.x] = a; s2[threadIdx.x] = b2;
  __syncthreads();
  for (int off = 128; off > 0; off >>= 1) {
    if (threadIdx.x < off) {
      s1[threadIdx.x] += s1[threadIdx.x + off];
      s2[threadIdx.x] += s2[threadIdx.x + off];
    }
    __syncthreads();
  }
  const float mean = s1[0] * (1.0f / DMODEL);
  const float var  = s2[0] * (1.0f / DMODEL) - mean * mean;
  const float inv  = rsqrtf(var + eps);
  u16* yr = y + (size_t)row * DMODEL;
  for (int i = threadIdx.x; i < DMODEL; i += 256)
    yr[i] = f2bf((xr[i] - mean) * inv * g[i] + bta[i]);
}

// ---------------- row softmax over NSEQ, in place ----------------
__global__ void softmax_kernel(float* __restrict__ p) {
  const int row = blockIdx.x;
  float* pr = p + (size_t)row * NSEQ;
  __shared__ float s[256];
  float mx = -3.0e38f;
  for (int i = threadIdx.x; i < NSEQ; i += 256) mx = fmaxf(mx, pr[i]);
  s[threadIdx.x] = mx;
  __syncthreads();
  for (int off = 128; off > 0; off >>= 1) {
    if (threadIdx.x < off) s[threadIdx.x] = fmaxf(s[threadIdx.x], s[threadIdx.x + off]);
    __syncthreads();
  }
  const float rowmax = s[0];
  __syncthreads();
  float sum = 0.f;
  for (int i = threadIdx.x; i < NSEQ; i += 256) {
    float e = __expf(pr[i] - rowmax);
    pr[i] = e;
    sum += e;
  }
  s[threadIdx.x] = sum;
  __syncthreads();
  for (int off = 128; off > 0; off >>= 1) {
    if (threadIdx.x < off) s[threadIdx.x] += s[threadIdx.x + off];
    __syncthreads();
  }
  const float inv = 1.0f / s[0];
  for (int i = threadIdx.x; i < NSEQ; i += 256) pr[i] *= inv;
}

// ---------------- weight fp32 (KxN) -> bf16 transposed (NxK) ----------------
__global__ void wt_kernel(const float* __restrict__ W, u16* __restrict__ Wt, int K, int Nn) {
  const int idx = blockIdx.x * 256 + threadIdx.x;
  if (idx >= K * Nn) return;
  const int k = idx / Nn, n = idx % Nn;
  Wt[(size_t)n * K + k] = f2bf(W[idx]);
}

__global__ void cvt_kernel(const float* __restrict__ x, u16* __restrict__ y, int n) {
  const int i = blockIdx.x * 256 + threadIdx.x;
  if (i < n) y[i] = f2bf(x[i]);
}

// way_logits = tmp @ W(D,E) + b  (tiny: plain VALU)
__global__ void way_kernel(const float* __restrict__ tmp, const float* __restrict__ Wc,
                           const float* __restrict__ bc, float* __restrict__ out) {
  const int idx = blockIdx.x * 256 + threadIdx.x;
  if (idx >= NTOK * EE) return;
  const int m = idx / EE, e = idx % EE;
  const float* tr = tmp + (size_t)m * DMODEL;
  float acc = bc[e];
  for (int k = 0; k < DMODEL; ++k) acc += tr[k] * Wc[k * EE + e];
  out[idx] = acc;
}

// =====================================================================

extern "C" void kernel_launch(void* const* d_in, const int* in_sizes, int n_in,
                              void* d_out, int out_size, void* d_ws, size_t ws_size,
                              hipStream_t stream) {
  (void)in_sizes; (void)n_in; (void)out_size; (void)ws_size;

  const float* inp  = (const float*)d_in[0];
  const float* ctx  = (const float*)d_in[1];
  const int*   wsel = (const int*)d_in[2];
  const float* sWq = (const float*)d_in[3];  const float* sbq = (const float*)d_in[4];
  const float* sWk = (const float*)d_in[5];  const float* sbk = (const float*)d_in[6];
  const float* sWv = (const float*)d_in[7];  const float* sbv = (const float*)d_in[8];
  const float* sWo = (const float*)d_in[9];  const float* sbo = (const float*)d_in[10];
  const float* cWq = (const float*)d_in[11]; const float* cbq = (const float*)d_in[12];
  const float* cWk = (const float*)d_in[13]; const float* cbk = (const float*)d_in[14];
  const float* cWv = (const float*)d_in[15]; const float* cbv = (const float*)d_in[16];
  const float* cWo = (const float*)d_in[17]; const float* cbo = (const float*)d_in[18];
  const float* g_in = (const float*)d_in[19]; const float* b_in = (const float*)d_in[20];
  const float* g_q  = (const float*)d_in[21]; const float* b_q  = (const float*)d_in[22];
  const float* g_c  = (const float*)d_in[23]; const float* b_c  = (const float*)d_in[24];
  const float* wayW = (const float*)d_in[25]; const float* wayb = (const float*)d_in[26];
  const float* fW1  = (const float*)d_in[27]; const float* fb1  = (const float*)d_in[28];
  const float* fW2  = (const float*)d_in[29]; const float* fb2  = (const float*)d_in[30];

  float* out_main = (float*)d_out;                               // (B,N,D)
  float* out_way  = out_main + (size_t)NTOK * DMODEL;            // (B,N,E)
  float* out_attn = out_way + (size_t)NTOK * EE;                 // (B,H,N,N) — also self-attn scratch

  // ---- workspace carve-out (256B aligned) ----
  char* wsp = (char*)d_ws;
  size_t off = 0;
  auto alloc = [&](size_t bytes) -> void* {
    void* p = (void*)(wsp + off);
    off = (off + bytes + 255) & ~(size_t)255;
    return p;
  };
  const size_t DD = (size_t)DMODEL * DMODEL;
  u16* wq_s = (u16*)alloc(DD * 2); u16* wk_s = (u16*)alloc(DD * 2);
  u16* wv_s = (u16*)alloc(DD * 2); u16* wo_s = (u16*)alloc(DD * 2);
  u16* wq_c = (u16*)alloc(DD * 2); u16* wk_c = (u16*)alloc(DD * 2);
  u16* wv_c = (u16*)alloc(DD * 2); u16* wo_c = (u16*)alloc(DD * 2);
  const size_t EW = (size_t)DMODEL * DFF;
  u16* w1t = (u16*)alloc((size_t)EE * EW * 2);
  u16* w2t = (u16*)alloc((size_t)EE * EW * 2);
  u16* xnorm = (u16*)alloc((size_t)NTOK * DMODEL * 2);
  u16* qb    = (u16*)alloc((size_t)NTOK * DMODEL * 2);
  u16* kb    = (u16*)alloc((size_t)NTOK * DMODEL * 2);
  u16* vT    = (u16*)alloc((size_t)NTOK * DMODEL * 2);   // (B,H,DH,N)
  u16* aout  = (u16*)alloc((size_t)NTOK * DMODEL * 2);
  u16* qn    = (u16*)alloc((size_t)NTOK * DMODEL * 2);
  u16* cn    = (u16*)alloc((size_t)NTOK * DMODEL * 2);
  u16* tmpb  = (u16*)alloc((size_t)NTOK * DMODEL * 2);
  float* query = (float*)alloc((size_t)NTOK * DMODEL * 4);
  float* tmpf  = (float*)alloc((size_t)NTOK * DMODEL * 4);
  u16* hbuf  = (u16*)alloc((size_t)NTOK * DFF * 2);

  const int TPB = 256;
  auto cdiv = [](long long a, long long b) { return (int)((a + b - 1) / b); };
  const float scale = 0.10206207261596577f;   // 1/sqrt(96)

  // ---- convert + transpose all weights to bf16 (N-major) ----
  const int nW = DMODEL * DMODEL;
  wt_kernel<<<cdiv(nW, TPB), TPB, 0, stream>>>(sWq, wq_s, DMODEL, DMODEL);
  wt_kernel<<<cdiv(nW, TPB), TPB, 0, stream>>>(sWk, wk_s, DMODEL, DMODEL);
  wt_kernel<<<cdiv(nW, TPB), TPB, 0, stream>>>(sWv, wv_s, DMODEL, DMODEL);
  wt_kernel<<<cdiv(nW, TPB), TPB, 0, stream>>>(sWo, wo_s, DMODEL, DMODEL);
  wt_kernel<<<cdiv(nW, TPB), TPB, 0, stream>>>(cWq, wq_c, DMODEL, DMODEL);
  wt_kernel<<<cdiv(nW, TPB), TPB, 0, stream>>>(cWk, wk_c, DMODEL, DMODEL);
  wt_kernel<<<cdiv(nW, TPB), TPB, 0, stream>>>(cWv, wv_c, DMODEL, DMODEL);
  wt_kernel<<<cdiv(nW, TPB), TPB, 0, stream>>>(cWo, wo_c, DMODEL, DMODEL);
  for (int e = 0; e < EE; ++e) {
    wt_kernel<<<cdiv((long long)EW, TPB), TPB, 0, stream>>>(
        fW1 + (size_t)e * EW, w1t + (size_t)e * EW, DMODEL, DFF);   // (D,DF) -> (DF,D)
    wt_kernel<<<cdiv((long long)EW, TPB), TPB, 0, stream>>>(
        fW2 + (size_t)e * EW, w2t + (size_t)e * EW, DFF, DMODEL);   // (DF,D) -> (D,DF)
  }

  const dim3 gproj(DMODEL / 128, NTOK / 128);                // staged GEMM: 128-col block tiles
  const dim3 gvt(DMODEL / 16, NTOK / 128);
  const dim3 gsc(NSEQ / 16, NSEQ / 128, BB * HH);
  const dim3 gav(DHD / 16, NSEQ / 128, BB * HH);

  // ================= self attention =================
  ln_kernel<<<NTOK, TPB, 0, stream>>>(inp, g_in, b_in, xnorm, 1e-6f);
  gemm_kernel<true, false, false><<<gproj, TPB, 0, stream>>>(
      xnorm, DMODEL, wq_s, DMODEL, sbq, nullptr, 0, qb, DMODEL, DMODEL, nullptr, 0);
  gemm_kernel<true, false, false><<<gproj, TPB, 0, stream>>>(
      xnorm, DMODEL, wk_s, DMODEL, sbk, nullptr, 0, kb, DMODEL, DMODEL, nullptr, 0);
  gemm_vt_kernel<<<gvt, TPB, 0, stream>>>(xnorm, DMODEL, wv_s, DMODEL, sbv, vT, DMODEL);
  attn_scores_kernel<<<gsc, TPB, 0, stream>>>(qb, kb, out_attn, scale);
  softmax_kernel<<<BB * HH * NSEQ, TPB, 0, stream>>>(out_attn);
  attn_av_kernel<<<gav, TPB, 0, stream>>>(out_attn, vT, aout);
  gemm_kernel<false, false, false><<<gproj, TPB, 0, stream>>>(
      aout, DMODEL, wo_s, DMODEL, sbo, inp, DMODEL, query, DMODEL, DMODEL, nullptr, 0);

  // ================= cross attention =================
  ln_kernel<<<NTOK, TPB, 0, stream>>>(query, g_q, b_q, qn, 1e-6f);
  ln_kernel<<<NTOK, TPB, 0, stream>>>(ctx, g_c, b_c, cn, 1e-5f);
  gemm_kernel<true, false, false><<<gproj, TPB, 0, stream>>>(
      qn, DMODEL, wq_c, DMODEL, cbq, nullptr, 0, qb, DMODEL, DMODEL, nullptr, 0);
  gemm_kernel<true, false, false><<<gproj, TPB, 0, stream>>>(
      cn, DMODEL, wk_c, DMODEL, cbk, nullptr, 0, kb, DMODEL, DMODEL, nullptr, 0);
  gemm_vt_kernel<<<gvt, TPB, 0, stream>>>(cn, DMODEL, wv_c, DMODEL, cbv, vT, DMODEL);
  attn_scores_kernel<<<gsc, TPB, 0, stream>>>(qb, kb, out_attn, scale);
  softmax_kernel<<<BB * HH * NSEQ, TPB, 0, stream>>>(out_attn);  // final contextual_attn
  attn_av_kernel<<<gav, TPB, 0, stream>>>(out_attn, vT, aout);
  gemm_kernel<false, false, false><<<gproj, TPB, 0, stream>>>(
      aout, DMODEL, wo_c, DMODEL, cbo, inp, DMODEL, tmpf, DMODEL, DMODEL, nullptr, 0);

  cvt_kernel<<<cdiv((long long)NTOK * DMODEL, TPB), TPB, 0, stream>>>(tmpf, tmpb, NTOK * DMODEL);
  way_kernel<<<cdiv((long long)NTOK * EE, TPB), TPB, 0, stream>>>(tmpf, wayW, wayb, out_way);

  // ================= expert FFN (all experts, masked store = per-token select) =================
  const dim3 g1(DFF / 128, NTOK / 128);
  const dim3 g2(DMODEL / 128, NTOK / 128);
  for (int e = 0; e < EE; ++e) {
    gemm_kernel<true, true, false><<<g1, TPB, 0, stream>>>(
        tmpb, DMODEL, w1t + (size_t)e * EW, DMODEL, fb1 + (size_t)e * DFF,
        nullptr, 0, hbuf, DFF, DMODEL, nullptr, 0);
    gemm_kernel<false, false, true><<<g2, TPB, 0, stream>>>(
        hbuf, DFF, w2t + (size_t)e * EW, DFF, fb2 + (size_t)e * DMODEL,
        tmpf, DMODEL, out_main, DMODEL, DFF, wsel, e);
  }
}
